// Swin3DTransformerBlock_65678639891290
// MI455X (gfx1250) — matrix operand order repaired and checked by reading, not compiled, not run on hardware
//
#include <hip/hip_runtime.h>
#include <hip/hip_bf16.h>

// ---------- types ----------
typedef __attribute__((ext_vector_type(16))) __bf16 v16bf;
typedef __attribute__((ext_vector_type(8)))  float  v8f;

union FragU { v16bf v; uint4 q[2]; };

__device__ inline unsigned short f2bf(float f) {
  unsigned u = __builtin_bit_cast(unsigned, f);
  u += 0x7FFFu + ((u >> 16) & 1u);           // round-to-nearest-even
  return (unsigned short)(u >> 16);
}

__device__ inline v8f wmma_bf16(v16bf a, v16bf b, v8f c) {
  return __builtin_amdgcn_wmma_f32_16x16x32_bf16(false, a, false, b, (short)0, c, false, false);
}

// ---------- CDNA5 async global->LDS DMA (ASYNCcnt-tracked) ----------
// LDS destination address is taken from the vdst VGPR; low 32 bits of the generic shared pointer
// are the wave-relative LDS byte offset (flat LDS aperture, ISA 10.2).
__device__ inline void async_g2l_b128(void* lds_dst, const void* gsrc) {
  unsigned ldsoff = (unsigned)(uintptr_t)lds_dst;
  unsigned long long ga = (unsigned long long)(uintptr_t)gsrc;
  asm volatile("global_load_async_to_lds_b128 %0, %1, off"
               :: "v"(ldsoff), "v"(ga) : "memory");
}
__device__ inline void async_wait0() {
  asm volatile("s_wait_asynccnt 0x0" ::: "memory");
}
__device__ inline void async_wait1() {   // allow 1 in flight (double buffering)
  asm volatile("s_wait_asynccnt 0x1" ::: "memory");
}

// ---------- constants ----------
#define CDIM   512
#define NHEADS 8
#define HD     64
#define SQ     144
#define NWIN   512      // B * nw = 2 * 256
#define TOK    73728    // B*Z*H*W
#define HID    2048

// ---------- kernel: fp32 -> bf16 elementwise ----------
__global__ __launch_bounds__(256) void cvt_bf16_kernel(const float* __restrict__ src,
                                                       unsigned short* __restrict__ dst, int n) {
  int i = blockIdx.x * 256 + threadIdx.x;
  if (i < n) dst[i] = f2bf(src[i]);
}

// ---------- kernel: earth position bias gather ----------
// ebias[(lev*8+lat)*8 + h][i][j], 256 maps of 144x144
__global__ __launch_bounds__(256) void epb_kernel(const float* __restrict__ table,
                                                  float* __restrict__ ebias) {
  size_t tid = (size_t)blockIdx.x * 256 + threadIdx.x;
  int j = (int)(tid % SQ);
  size_t t = tid / SQ;
  int i = (int)(t % SQ);
  int mh = (int)(t / SQ);          // 0..255
  int h   = mh & 7;
  int map = mh >> 3;               // lev*8+lat
  int lat = map & 7, lev = map >> 3;
  int zi = i / 72, ri = i % 72, hi = ri / 12, wi = ri % 12;
  int zj = j / 72, rj = j % 72, hj = rj / 12, wj = rj % 12;
  int d0 = zi + 2 * zj;            // [0,3]
  int d1 = hi + 6 * hj;            // [0,35]
  int d2 = wi - wj + 11;           // [0,22]
  int idx = d0 * 828 + d1 * 23 + d2;
  ebias[tid] = table[(((size_t)idx * 4 + lev) * 8 + lat) * 8 + h];
}

// ---------- kernel: LayerNorm1 + 3D window partition -> bf16 ----------
__global__ __launch_bounds__(256) void ln1_part_kernel(const float* __restrict__ x,
                                                       const float* __restrict__ gw,
                                                       const float* __restrict__ gb,
                                                       unsigned short* __restrict__ xw) {
  int token = blockIdx.x * 8 + (threadIdx.x >> 5);
  int lane  = threadIdx.x & 31;
  const float* row = x + (size_t)token * CDIM;
  float vals[16];
  float s = 0.f, ss = 0.f;
#pragma unroll
  for (int i = 0; i < 16; ++i) { float v = row[lane + i * 32]; vals[i] = v; s += v; ss += v * v; }
#pragma unroll
  for (int off = 16; off > 0; off >>= 1) { s += __shfl_xor(s, off, 32); ss += __shfl_xor(ss, off, 32); }
  float mu   = s * (1.f / CDIM);
  float var  = ss * (1.f / CDIM) - mu * mu;
  float rstd = rsqrtf(var + 1e-5f);
  // token -> (b,z,h,w) -> (win, tok-in-window)
  int ww = token % 96; int t1 = token / 96;
  int hh = t1 % 48;    int t2 = t1 / 48;
  int zz = t2 % 8;     int bb = t2 / 8;
  int win = ((bb * 4 + (zz >> 1)) * 8 + hh / 6) * 8 + ww / 12;
  int tk  = (zz & 1) * 72 + (hh % 6) * 12 + (ww % 12);
  unsigned short* orow = xw + ((size_t)win * SQ + tk) * CDIM;
#pragma unroll
  for (int i = 0; i < 16; ++i) {
    int c = lane + i * 32;
    orow[c] = f2bf((vals[i] - mu) * rstd * gw[c] + gb[c]);
  }
}

// ---------- kernel: LayerNorm2 (token-major in/out) ----------
__global__ __launch_bounds__(256) void ln2_kernel(const float* __restrict__ x,
                                                  const float* __restrict__ gw,
                                                  const float* __restrict__ gb,
                                                  unsigned short* __restrict__ out) {
  int token = blockIdx.x * 8 + (threadIdx.x >> 5);
  int lane  = threadIdx.x & 31;
  const float* row = x + (size_t)token * CDIM;
  float vals[16];
  float s = 0.f, ss = 0.f;
#pragma unroll
  for (int i = 0; i < 16; ++i) { float v = row[lane + i * 32]; vals[i] = v; s += v; ss += v * v; }
#pragma unroll
  for (int off = 16; off > 0; off >>= 1) { s += __shfl_xor(s, off, 32); ss += __shfl_xor(ss, off, 32); }
  float mu   = s * (1.f / CDIM);
  float var  = ss * (1.f / CDIM) - mu * mu;
  float rstd = rsqrtf(var + 1e-5f);
  unsigned short* orow = out + (size_t)token * CDIM;
#pragma unroll
  for (int i = 0; i < 16; ++i) {
    int c = lane + i * 32;
    orow[c] = f2bf((vals[i] - mu) * rstd * gw[c] + gb[c]);
  }
}

// ---------- generic bf16 WMMA GEMM: C[M x Nout] = A[M x K] * Wt[Nout x K]^T ----------
// Block: 256 threads = 8 waves. Tile: 128(M) x 64(N). K step 32.
// B tile double-buffered in LDS via async global->LDS DMA: DMA of tile k+1
// overlaps WMMA on tile k; s_wait_asynccnt 1 retires tile k (in-order).
// MODE 0: QKV split (q scaled, v transposed)   MODE 1: proj + window-reverse + residual
// MODE 2: fc1 + GELU(exact)                    MODE 3: fc2 + residual -> fp32 out
template <int MODE>
__global__ __launch_bounds__(256)
void gemm_bf16(const unsigned short* __restrict__ A,
               const unsigned short* __restrict__ Wt,
               const float* __restrict__ bias,
               int Kdim, int Nout, float scale,
               unsigned short* __restrict__ q_out,
               unsigned short* __restrict__ k_out,
               unsigned short* __restrict__ v_out,
               const float* __restrict__ resid,
               float* __restrict__ out_f32,
               unsigned short* __restrict__ out_bf16) {
  __shared__ uint4 smemB[640];                 // 2 buffers x (64 rows x 80 bytes)
  unsigned char* sB = (unsigned char*)smemB;

  const int tid  = threadIdx.x;
  const int lane = tid & 31;
  const int wv   = tid >> 5;
  const int M0   = blockIdx.x * 128 + wv * 16;
  const int N0   = blockIdx.y * 64;

  v8f acc0 = {0.f,0.f,0.f,0.f,0.f,0.f,0.f,0.f};
  v8f acc1 = acc0, acc2 = acc0, acc3 = acc0;

  const unsigned short* Arow = A + (size_t)(M0 + (lane & 15)) * Kdim + ((lane >> 4) << 3);
  const int br = tid >> 2;                      // B-tile row 0..63
  const int bq = tid & 3;                       // uint4 within row
  const unsigned short* Wrow = Wt + (size_t)(N0 + br) * Kdim + bq * 8;
  unsigned char* myDst = sB + br * 80 + bq * 16;

  const int nIter = Kdim >> 5;
  // prologue: DMA tile 0 into buffer 0
  async_g2l_b128(myDst, Wrow);

  for (int i = 0; i < nIter; ++i) {
    const int k0 = i << 5;
    __syncthreads();                            // everyone done reading buf[(i+1)&1]
    if (i + 1 < nIter) {
      async_g2l_b128(myDst + ((i + 1) & 1) * 5120, Wrow + k0 + 32);
      async_wait1();                            // tile i landed; tile i+1 in flight
    } else {
      async_wait0();
    }
    __syncthreads();

    FragU fa;
    fa.q[0] = *(const uint4*)(Arow + k0);
    fa.q[1] = *(const uint4*)(Arow + k0 + 16);
    __builtin_prefetch(Arow + k0 + 64, 0, 0);   // global_prefetch_b8 on streamed A

    const unsigned char* bbase = sB + (i & 1) * 5120 + (lane & 15) * 80 + ((lane >> 4) << 4);
    FragU fb;
#pragma unroll
    for (int nt = 0; nt < 4; ++nt) {
      const unsigned char* bp = bbase + nt * 16 * 80;
      fb.q[0] = *(const uint4*)(bp);
      fb.q[1] = *(const uint4*)(bp + 32);
      if (nt == 0) acc0 = wmma_bf16(fa.v, fb.v, acc0);
      if (nt == 1) acc1 = wmma_bf16(fa.v, fb.v, acc1);
      if (nt == 2) acc2 = wmma_bf16(fa.v, fb.v, acc2);
      if (nt == 3) acc3 = wmma_bf16(fa.v, fb.v, acc3);
    }
  }

  // ---- epilogue ----
  v8f accs[4] = {acc0, acc1, acc2, acc3};
#pragma unroll
  for (int nt = 0; nt < 4; ++nt) {
    int n = N0 + nt * 16 + (lane & 15);
    float bs = bias[n];
#pragma unroll
    for (int j = 0; j < 8; ++j) {
      int m = M0 + j + ((lane >> 4) << 3);
      float val = accs[nt][j] + bs;
      if (MODE == 0) {
        int which = n >> 9, h = (n >> 6) & 7, d = n & 63;
        int win = m / SQ, tk = m - win * SQ;
        if (which == 0)
          q_out[(((size_t)(win * NHEADS + h)) * SQ + tk) * HD + d] = f2bf(val * scale);
        else if (which == 1)
          k_out[(((size_t)(win * NHEADS + h)) * SQ + tk) * HD + d] = f2bf(val);
        else
          v_out[(((size_t)(win * NHEADS + h)) * HD + d) * SQ + tk] = f2bf(val);
      } else if (MODE == 1) {
        int win = m / SQ, tk = m - win * SQ;
        int b = win >> 8, wl = win & 255;
        int lev = wl >> 6, lat = (wl >> 3) & 7, lon = wl & 7;
        int iz = tk / 72, rem = tk - iz * 72, ih = rem / 12, iw = rem - ih * 12;
        size_t t = (((size_t)b * 8 + (lev * 2 + iz)) * 48 + (lat * 6 + ih)) * 96 + (lon * 12 + iw);
        out_f32[t * CDIM + n] = resid[t * CDIM + n] + val;
      } else if (MODE == 2) {
        float g = 0.5f * val * (1.0f + erff(val * 0.70710678f));
        out_bf16[(size_t)m * Nout + n] = f2bf(g);
      } else {
        out_f32[(size_t)m * CDIM + n] = resid[(size_t)m * CDIM + n] + val;
      }
    }
  }
}

// ---------- fused windowed attention: S = q k^T (+bias) -> softmax -> P V ----------
// one block per (window, head); entire S kept in LDS; staging via async DMA
__global__ __launch_bounds__(256)
void attn_kernel(const unsigned short* __restrict__ qg,
                 const unsigned short* __restrict__ kg,
                 const unsigned short* __restrict__ vtg,
                 const float* __restrict__ ebias,
                 unsigned short* __restrict__ og) {
  extern __shared__ unsigned char smem[];
  unsigned short* sq  = (unsigned short*)smem;         // 144 x 64
  unsigned short* sk  = sq + SQ * HD;                  // 144 x 64
  unsigned short* svt = sk + SQ * HD;                  // 64 x 160 (K padded)
  float*          sS  = (float*)(svt + HD * 160);      // 144 x 148 (padded stride)
  unsigned short* sP  = (unsigned short*)(sS + SQ * 148); // 144 x 160

  const int tid = threadIdx.x, lane = tid & 31, wv = tid >> 5;
  const int wh  = blockIdx.x;           // 0..4095
  const int win = wh >> 3, h = wh & 7;
  const size_t base = (size_t)wh * SQ * HD;

  // stage q, k (flat 18KB each), vT (64x144 -> 64x160 with zero pad): async DMA
  for (int e = tid; e < 1152; e += 256) {
    async_g2l_b128(sq + e * 8, qg + base + e * 8);
    async_g2l_b128(sk + e * 8, kg + base + e * 8);
  }
  for (int e = tid; e < 1152; e += 256) {
    int r = e / 18, c = e % 18;
    async_g2l_b128(svt + r * 160 + c * 8, vtg + base + e * 8);
  }
  if (tid < 128) {
    uint4 z4 = make_uint4(0u, 0u, 0u, 0u);
    int r = tid >> 1, c = tid & 1;
    *(uint4*)(svt + r * 160 + SQ + c * 8) = z4;
  }
  async_wait0();
  __syncthreads();

  // ---- S = q k^T + bias : 9x9 tiles of 16x16, K = 64 ----
  const float* bias = ebias + (size_t)(((win & 255) >> 3) * 8 + h) * SQ * SQ;
  for (int t = wv; t < 81; t += 8) {
    int M0 = (t / 9) * 16, N0 = (t % 9) * 16;
    v8f c = {0.f,0.f,0.f,0.f,0.f,0.f,0.f,0.f};
    const unsigned short* ap = sq + (M0 + (lane & 15)) * HD + ((lane >> 4) << 3);
    const unsigned short* bp = sk + (N0 + (lane & 15)) * HD + ((lane >> 4) << 3);
#pragma unroll
    for (int k0 = 0; k0 < HD; k0 += 32) {
      FragU fa, fb;
      fa.q[0] = *(const uint4*)(ap + k0);
      fa.q[1] = *(const uint4*)(ap + k0 + 16);
      fb.q[0] = *(const uint4*)(bp + k0);
      fb.q[1] = *(const uint4*)(bp + k0 + 16);
      c = wmma_bf16(fa.v, fb.v, c);
    }
    int N  = N0 + (lane & 15);
    int Mb = M0 + ((lane >> 4) << 3);
#pragma unroll
    for (int j = 0; j < 8; ++j)
      sS[(Mb + j) * 148 + N] = c[j] + bias[(Mb + j) * SQ + N];
  }
  __syncthreads();

  // ---- row softmax (float4 passes) -> bf16 P (K padded to 160 with zeros) ----
  if (tid < SQ) {
    float4* row4 = (float4*)(sS + tid * 148);   // 592B row stride, 16B aligned
    float mx = -3.4e38f;
#pragma unroll 4
    for (int j = 0; j < 36; ++j) {
      float4 v = row4[j];
      mx = fmaxf(mx, fmaxf(fmaxf(v.x, v.y), fmaxf(v.z, v.w)));
    }
    float sum = 0.f;
#pragma unroll 4
    for (int j = 0; j < 36; ++j) {
      float4 v = row4[j];
      v.x = __expf(v.x - mx); v.y = __expf(v.y - mx);
      v.z = __expf(v.z - mx); v.w = __expf(v.w - mx);
      row4[j] = v;
      sum += (v.x + v.y) + (v.z + v.w);
    }
    float inv = 1.f / sum;
    unsigned* pr32 = (unsigned*)(sP + tid * 160);
#pragma unroll 4
    for (int j = 0; j < 36; ++j) {
      float4 v = row4[j];
      pr32[j * 2 + 0] = (unsigned)f2bf(v.x * inv) | ((unsigned)f2bf(v.y * inv) << 16);
      pr32[j * 2 + 1] = (unsigned)f2bf(v.z * inv) | ((unsigned)f2bf(v.w * inv) << 16);
    }
    uint4 z4 = make_uint4(0u, 0u, 0u, 0u);
    *(uint4*)(sP + tid * 160 + SQ) = z4;
    *(uint4*)(sP + tid * 160 + SQ + 8) = z4;
  }
  __syncthreads();

  // ---- O = P V : 9x4 tiles of 16x16, K = 160 ----
  unsigned short* orow = og + (size_t)win * SQ * CDIM + h * HD;
  for (int t = wv; t < 36; t += 8) {
    int M0 = (t >> 2) * 16, N0 = (t & 3) * 16;
    v8f c = {0.f,0.f,0.f,0.f,0.f,0.f,0.f,0.f};
    const unsigned short* ap = sP  + (M0 + (lane & 15)) * 160 + ((lane >> 4) << 3);
    const unsigned short* bp = svt + (N0 + (lane & 15)) * 160 + ((lane >> 4) << 3);
#pragma unroll
    for (int k0 = 0; k0 < 160; k0 += 32) {
      FragU fa, fb;
      fa.q[0] = *(const uint4*)(ap + k0);
      fa.q[1] = *(const uint4*)(ap + k0 + 16);
      fb.q[0] = *(const uint4*)(bp + k0);
      fb.q[1] = *(const uint4*)(bp + k0 + 16);
      c = wmma_bf16(fa.v, fb.v, c);
    }
    int N  = N0 + (lane & 15);
    int Mb = M0 + ((lane >> 4) << 3);
#pragma unroll
    for (int j = 0; j < 8; ++j)
      orow[(size_t)(Mb + j) * CDIM + N] = f2bf(c[j]);
  }
}

// ---------- host launcher ----------
extern "C" void kernel_launch(void* const* d_in, const int* in_sizes, int n_in,
                              void* d_out, int out_size, void* d_ws, size_t ws_size,
                              hipStream_t stream) {
  (void)in_sizes; (void)n_in; (void)out_size; (void)ws_size;
  const float* x      = (const float*)d_in[0];
  const float* n1w    = (const float*)d_in[1];
  const float* n1b    = (const float*)d_in[2];
  const float* qkv_w  = (const float*)d_in[3];
  const float* qkv_b  = (const float*)d_in[4];
  const float* proj_w = (const float*)d_in[5];
  const float* proj_b = (const float*)d_in[6];
  const float* table  = (const float*)d_in[7];
  const float* n2w    = (const float*)d_in[8];
  const float* n2b    = (const float*)d_in[9];
  const float* fc1_w  = (const float*)d_in[10];
  const float* fc1_b  = (const float*)d_in[11];
  const float* fc2_w  = (const float*)d_in[12];
  const float* fc2_b  = (const float*)d_in[13];
  float* out = (float*)d_out;

  size_t off = 0;
  char* wsb = (char*)d_ws;
  auto carve = [&](size_t bytes) -> void* {
    void* p = wsb + off;
    off += (bytes + 255) & ~(size_t)255;
    return p;
  };
  unsigned short* wqkv = (unsigned short*)carve((size_t)1536 * 512 * 2);
  unsigned short* wprj = (unsigned short*)carve((size_t)512 * 512 * 2);
  unsigned short* wf1  = (unsigned short*)carve((size_t)2048 * 512 * 2);
  unsigned short* wf2  = (unsigned short*)carve((size_t)512 * 2048 * 2);
  float*          ebias= (float*)carve((size_t)256 * SQ * SQ * 4);
  unsigned short* xw   = (unsigned short*)carve((size_t)NWIN * SQ * CDIM * 2);
  unsigned short* qg   = (unsigned short*)carve((size_t)NWIN * SQ * CDIM * 2);
  unsigned short* kg   = (unsigned short*)carve((size_t)NWIN * SQ * CDIM * 2);
  unsigned short* vtg  = (unsigned short*)carve((size_t)NWIN * SQ * CDIM * 2);
  unsigned short* og   = (unsigned short*)carve((size_t)NWIN * SQ * CDIM * 2);
  float*          xatt = (float*)carve((size_t)TOK * CDIM * 4);
  unsigned short* hln2 = (unsigned short*)carve((size_t)TOK * CDIM * 2);
  unsigned short* h1   = (unsigned short*)carve((size_t)TOK * HID * 2);

  // 1) weights -> bf16
  cvt_bf16_kernel<<<(1536 * 512) / 256, 256, 0, stream>>>(qkv_w, wqkv, 1536 * 512);
  cvt_bf16_kernel<<<(512 * 512) / 256, 256, 0, stream>>>(proj_w, wprj, 512 * 512);
  cvt_bf16_kernel<<<(2048 * 512) / 256, 256, 0, stream>>>(fc1_w, wf1, 2048 * 512);
  cvt_bf16_kernel<<<(512 * 2048) / 256, 256, 0, stream>>>(fc2_w, wf2, 512 * 2048);

  // 2) earth position bias gather
  epb_kernel<<<(256 * SQ * SQ) / 256, 256, 0, stream>>>(table, ebias);

  // 3) LN1 + window partition
  ln1_part_kernel<<<TOK / 8, 256, 0, stream>>>(x, n1w, n1b, xw);

  // 4) QKV GEMM
  gemm_bf16<0><<<dim3(TOK / 128, 1536 / 64), 256, 0, stream>>>(
      xw, wqkv, qkv_b, 512, 1536, 0.125f, qg, kg, vtg, nullptr, nullptr, nullptr);

  // 5) fused attention (LDS: q+k+vT+S+P = 188672 bytes)
  size_t attn_lds = (size_t)SQ * HD * 2 * 2 + (size_t)HD * 160 * 2 +
                    (size_t)SQ * 148 * 4 + (size_t)SQ * 160 * 2;
  attn_kernel<<<NWIN * NHEADS, 256, attn_lds, stream>>>(qg, kg, vtg, ebias, og);

  // 6) proj GEMM + window reverse + residual
  gemm_bf16<1><<<dim3(TOK / 128, 512 / 64), 256, 0, stream>>>(
      og, wprj, proj_b, 512, 512, 1.f, nullptr, nullptr, nullptr, x, xatt, nullptr);

  // 7) LN2
  ln2_kernel<<<TOK / 8, 256, 0, stream>>>(xatt, n2w, n2b, hln2);

  // 8) fc1 + GELU
  gemm_bf16<2><<<dim3(TOK / 128, 2048 / 64), 256, 0, stream>>>(
      hln2, wf1, fc1_b, 512, 2048, 1.f, nullptr, nullptr, nullptr, nullptr, nullptr, h1);

  // 9) fc2 + residual -> out
  gemm_bf16<3><<<dim3(TOK / 128, 512 / 64), 256, 0, stream>>>(
      h1, wf2, fc2_b, 2048, 512, 1.f, nullptr, nullptr, nullptr, xatt, out, nullptr);
}
